// MembershipDecoder_7275674599528
// MI455X (gfx1250) — compile-verified, matrix operand order checked by reading, not run on hardware
//
#include <hip/hip_runtime.h>
#include <hip/hip_bf16.h>

typedef __attribute__((ext_vector_type(16))) _Float16 v16h;
typedef __attribute__((ext_vector_type(8)))  _Float16 v8h;
typedef __attribute__((ext_vector_type(8)))  float    v8f;

#define LN_EPS 1e-3f
#define HDIM 32
#define SLOTS_CAP 8192   // num_seq(32) * seq_len(256) from setup

// ---------------------------------------------------------------------------
// K0/K1: per-row projection.  x -> s = relu(x@W1 + b1) -> a = s@W2 (+ bd)
// Emits a as f16 (WMMA operand) + per-row scalars Au = a·u, ma = mean(a),
// Sa = sum(a^2).  One wave32 per row (8 rows per 256-thread block).
// ---------------------------------------------------------------------------
__global__ __launch_bounds__(256) void md_proj_kernel(
    const float* __restrict__ X, int n, int d,
    const float* __restrict__ W1, const float* __restrict__ b1,
    const float* __restrict__ W2, const float* __restrict__ bd,  // bd==null for sites
    const float* __restrict__ ln_g, const float* __restrict__ W_o,
    _Float16* __restrict__ outF16, float* __restrict__ outU,
    float* __restrict__ outM, float* __restrict__ outS)
{
    __shared__ float sh[8][HDIM];
    const int tid = threadIdx.x;
    const int ls = tid >> 5;          // local row (one wave32 per row)
    const int k  = tid & 31;          // hidden index 0..31
    const int i  = blockIdx.x * 8 + ls;
    if (i < n) {
        const float* x = X + (size_t)i * d;
        float s = b1[k];
        for (int t = 0; t < d; ++t) s = fmaf(x[t], W1[t * HDIM + k], s);
        sh[ls][k] = fmaxf(s, 0.f);
    }
    __syncthreads();
    if (i >= n) return;
    float a = bd ? bd[k] : 0.f;
    for (int h = 0; h < HDIM; ++h) a = fmaf(sh[ls][h], W2[h * HDIM + k], a);
    const float u = ln_g[k] * W_o[k];
    float pu = a * u, pm = a, ps = a * a;
    for (int m = 16; m; m >>= 1) {
        pu += __shfl_xor(pu, m, 32);
        pm += __shfl_xor(pm, m, 32);
        ps += __shfl_xor(ps, m, 32);
    }
    outF16[(size_t)i * HDIM + k] = (_Float16)a;
    if (k == 0) { outU[i] = pu; outM[i] = pm * (1.f / 32.f); outS[i] = ps; }
}

// ---------------------------------------------------------------------------
// K2 (fused): per 16-row block — WMMA logits -> LDS -> softmax -> M (one HBM
// write) -> bidirectional cumsum scan -> scatter atomics.  Row block of
// 16 x nc f32 logits (64 KB for nc=1024) stays resident in LDS (320 KB/WGP).
//
// Phase A: 8 waves x (nc/16/8) tiles, one v_wmma_f32_16x16x32_f16 per tile
//   (K=32 in one instruction).  Per-lane operand gathers follow the 16-bit
//   A/B VGPR layouts: lane L (r=L&15, hi=L>>4): halves 0..7 = K(hi*8..+7),
//   halves 8..15 = K(16+hi*8..+7).  D: elem v at lane L -> (M=v+8*hi, N=r).
// Phase B: wave w owns rows 2w, 2w+1: softmax via __shfl_xor reductions.
// Phase C: same wave: inclusive wave scan (__shfl_up) gives cumsum AND
//   reverse cumsum (total - excl), scatter-added with f32 atomics.
// ---------------------------------------------------------------------------
__global__ __launch_bounds__(256) void md_fused_rowblock_kernel(
    const _Float16* __restrict__ aF16, const _Float16* __restrict__ bF16,
    const float* __restrict__ siteAu, const float* __restrict__ siteMa,
    const float* __restrict__ siteSa,
    const float* __restrict__ conBu,  const float* __restrict__ conMb,
    const float* __restrict__ conSb,
    const float* __restrict__ ln_g, const float* __restrict__ ln_b,
    const float* __restrict__ W_o,  const float* __restrict__ b_o,
    const int* __restrict__ idxL, const int* __restrict__ idxH,
    float* __restrict__ Mout, float* __restrict__ Zl, float* __restrict__ Zh,
    int ns, int nc)
{
    extern __shared__ float Lrow[];              // [16][nc] logits row block
    const int rowBase = blockIdx.x * 16;
    const int tid  = threadIdx.x;
    const int w    = tid >> 5;                   // wave id 0..7
    const int lane = tid & 31;
    const int r  = lane & 15;
    const int hi = lane >> 4;

    // sumU and C0 via wave32 reduction (lane == hidden index, H == 32)
    float uk = ln_g[lane] * W_o[lane];
    float ck = ln_b[lane] * W_o[lane];
    for (int m = 16; m; m >>= 1) {
        uk += __shfl_xor(uk, m, 32);
        ck += __shfl_xor(ck, m, 32);
    }
    const float sumU = uk;
    const float C0 = ck + b_o[0];

    // ---- Phase A: WMMA tiles -> LDS --------------------------------------
    const _Float16* ap = aF16 + (size_t)(rowBase + r) * HDIM + hi * 8;
    v16h A;
    {
        v8h lo = *(const v8h*)(ap);
        v8h hh = *(const v8h*)(ap + 16);
        for (int t = 0; t < 8; ++t) { A[t] = lo[t]; A[t + 8] = hh[t]; }
    }
    // per-row scalars for the 8 output rows this lane produces
    const int numCT = nc >> 4;
    for (int ctile = w; ctile < numCT; ctile += 8) {
        const int colBase = ctile << 4;
        const _Float16* bp = bF16 + (size_t)(colBase + r) * HDIM + hi * 8;
        v16h B;
        {
            v8h lo = *(const v8h*)(bp);
            v8h hh = *(const v8h*)(bp + 16);
            for (int t = 0; t < 8; ++t) { B[t] = lo[t]; B[t + 8] = hh[t]; }
        }
        v8f C = {};
        C = __builtin_amdgcn_wmma_f32_16x16x32_f16(
                /*neg_a=*/false, A, /*neg_b=*/false, B,
                /*c_mod=*/(short)0, C, /*reuse_a=*/false, /*reuse_b=*/false);

        const int j = colBase + r;
        const float Bu = conBu[j], mb = conMb[j], Sb = conSb[j];
        for (int v = 0; v < 8; ++v) {
            const int m   = v + 8 * hi;          // row within tile
            const int i   = rowBase + m;
            const float Au = siteAu[i], ma = siteMa[i], Sa = siteSa[i];
            const float mu = ma + mb;
            const float var = (Sa + 2.f * C[v] + Sb) * (1.f / 32.f) - mu * mu;
            const float inv = rsqrtf(var + LN_EPS);
            Lrow[m * nc + j] = (Au + Bu - mu * sumU) * inv + C0;
        }
    }
    __syncthreads();

    // ---- Phase B+C: per-row softmax, store M, scan + scatter --------------
    const int cpl = nc >> 5;                     // contiguous cols per lane
    for (int mr = 0; mr < 2; ++mr) {
        const int m = 2 * w + mr;
        const int i = rowBase + m;
        if (i >= ns) continue;
        float* row = Lrow + m * nc;

        // softmax (wave-level)
        float mx = -3.402823466e38f;
        for (int j = lane; j < nc; j += 32) mx = fmaxf(mx, row[j]);
        for (int s = 16; s; s >>= 1) mx = fmaxf(mx, __shfl_xor(mx, s, 32));
        float sum = 0.f;
        for (int j = lane; j < nc; j += 32) {
            const float e = expf(row[j] - mx);
            row[j] = e;
            sum += e;
        }
        for (int s = 16; s; s >>= 1) sum += __shfl_xor(sum, s, 32);
        const float isum = 1.f / sum;
        float* mo = Mout + (size_t)i * nc;
        for (int j = lane; j < nc; j += 32) {
            const float v = row[j] * isum;
            row[j] = v;
            mo[j] = v;                           // the ONLY HBM write of M
        }

        // inclusive wave scan over per-lane chunk totals
        const int c0 = lane * cpl;
        float run = 0.f;
        for (int t = 0; t < cpl; ++t) run += row[c0 + t];
        float incl = run;
        for (int off = 1; off < 32; off <<= 1) {
            const float x = __shfl_up(incl, off, 32);
            if (lane >= off) incl += x;
        }
        const float total = __shfl(incl, 31, 32);
        float acc = incl - run;                  // exclusive base for this lane

        const size_t baseH = (size_t)idxH[i] * nc + c0;
        const size_t baseL = (size_t)idxL[i] * nc + c0;
        for (int t = 0; t < cpl; ++t) {
            const float v = row[c0 + t];
            acc += v;                            // inclusive cumsum  -> higher
            const float cl = total - acc + v;    // reverse cumsum    -> lower
            atomicAdd(&Zh[baseH + t], acc);
            atomicAdd(&Zl[baseL + t], cl);
        }
    }
}

// ---------------------------------------------------------------------------
// K3: zero the two Z scatter buffers (contiguous)
// ---------------------------------------------------------------------------
__global__ void md_zero_kernel(float* __restrict__ Z, size_t n)
{
    for (size_t p = (size_t)blockIdx.x * blockDim.x + threadIdx.x; p < n;
         p += (size_t)gridDim.x * blockDim.x)
        Z[p] = 0.f;
}

// ---------------------------------------------------------------------------
// K4: exclusive cumprod of relu(1-Z) along the seq_len axis, in place.
// 2 * num_seq * nc independent chains; adjacent threads walk adjacent columns
// so every step is a coalesced 128B line per wave.
// ---------------------------------------------------------------------------
__global__ void md_cumprod_kernel(float* __restrict__ Zl, float* __restrict__ Zh,
                                  const int* __restrict__ pnum,
                                  const int* __restrict__ plen, int nc)
{
    const int num = *pnum, len = *plen;
    const long perBuf = (long)num * nc;
    const long chains = 2L * perBuf;
    for (long c = (long)blockIdx.x * blockDim.x + threadIdx.x; c < chains;
         c += (long)gridDim.x * blockDim.x) {
        float* Z = (c < perBuf) ? Zl : Zh;
        const long cc = (c < perBuf) ? c : (c - perBuf);
        const int seq = (int)(cc / nc);
        const int col = (int)(cc % nc);
        float* p = Z + (size_t)seq * len * nc + col;
        float prod = 1.f;
        for (int s = 0; s < len; ++s) {
            if (s + 8 < len) __builtin_prefetch(p + (size_t)(s + 8) * nc, 0, 0);
            const float v = p[(size_t)s * nc];
            p[(size_t)s * nc] = prod;          // exclusive product written back
            prod *= fmaxf(1.f - v, 0.f);
        }
    }
}

// ---------------------------------------------------------------------------
// K5: gather bands per site and final elementwise product
// ---------------------------------------------------------------------------
__global__ __launch_bounds__(256) void md_final_kernel(
    const float* __restrict__ M, const float* __restrict__ Zl,
    const float* __restrict__ Zh, const int* __restrict__ idxL,
    const int* __restrict__ idxH, float* __restrict__ out, int ns, int nc)
{
    const long n = (long)ns * nc;
    for (long p = (long)blockIdx.x * blockDim.x + threadIdx.x; p < n;
         p += (long)gridDim.x * blockDim.x) {
        const int i = (int)(p / nc);
        const int j = (int)(p % nc);
        out[p] = M[p] * Zl[(size_t)idxL[i] * nc + j] * Zh[(size_t)idxH[i] * nc + j];
    }
}

// ---------------------------------------------------------------------------
extern "C" void kernel_launch(void* const* d_in, const int* in_sizes, int n_in,
                              void* d_out, int out_size, void* d_ws, size_t ws_size,
                              hipStream_t stream)
{
    // setup_inputs() order:
    const float* sites  = (const float*)d_in[0];
    const float* cons   = (const float*)d_in[1];
    const int*   idxL   = (const int*)d_in[2];
    const int*   idxH   = (const int*)d_in[3];
    const float* W_s    = (const float*)d_in[4];
    const float* b_s    = (const float*)d_in[5];
    const float* W_c    = (const float*)d_in[6];
    const float* b_c    = (const float*)d_in[7];
    const float* W_d    = (const float*)d_in[8];
    const float* b_d    = (const float*)d_in[9];
    const float* ln_g   = (const float*)d_in[10];
    const float* ln_b   = (const float*)d_in[11];
    const float* W_o    = (const float*)d_in[12];
    const float* b_o    = (const float*)d_in[13];
    const int*   pNum   = (const int*)d_in[14];   // device scalar num_seq
    const int*   pLen   = (const int*)d_in[15];   // device scalar seq_len

    const int d  = in_sizes[4] / HDIM;            // 64
    const int ns = in_sizes[0] / d;               // 4096
    const int nc = in_sizes[1] / d;               // 1024

    // workspace layout (256B aligned regions)
    char* ws = (char*)d_ws;
    size_t off = 0;
    auto alloc = [&](size_t bytes) -> void* {
        void* p = ws + off;
        off = (off + bytes + 255) & ~(size_t)255;
        return p;
    };
    _Float16* aF16  = (_Float16*)alloc((size_t)ns * HDIM * sizeof(_Float16));
    _Float16* bF16  = (_Float16*)alloc((size_t)nc * HDIM * sizeof(_Float16));
    float* siteAu   = (float*)alloc((size_t)ns * 4);
    float* siteMa   = (float*)alloc((size_t)ns * 4);
    float* siteSa   = (float*)alloc((size_t)ns * 4);
    float* conBu    = (float*)alloc((size_t)nc * 4);
    float* conMb    = (float*)alloc((size_t)nc * 4);
    float* conSb    = (float*)alloc((size_t)nc * 4);
    float* Mbuf     = (float*)alloc((size_t)ns * nc * 4);
    float* Zboth    = (float*)alloc(2ull * SLOTS_CAP * nc * 4);
    float* Zl = Zboth;
    float* Zh = Zboth + (size_t)SLOTS_CAP * nc;

    // 1: zero both scatter grids FIRST (fused kernel scatters into them)
    md_zero_kernel<<<2048, 256, 0, stream>>>(Zboth, 2ull * SLOTS_CAP * nc);

    // 2/3: projections (sites: no extra bias; cons: fold b_d)
    md_proj_kernel<<<(ns + 7) / 8, 256, 0, stream>>>(
        sites, ns, d, W_s, b_s, W_d, nullptr, ln_g, W_o,
        aF16, siteAu, siteMa, siteSa);
    md_proj_kernel<<<(nc + 7) / 8, 256, 0, stream>>>(
        cons, nc, d, W_c, b_c, W_d, b_d, ln_g, W_o,
        bF16, conBu, conMb, conSb);

    // 4: fused WMMA logits -> softmax -> M write -> scan -> scatter
    const size_t ldsBytes = (size_t)16 * nc * sizeof(float);
    md_fused_rowblock_kernel<<<ns / 16, 256, ldsBytes, stream>>>(
        aF16, bF16, siteAu, siteMa, siteSa, conBu, conMb, conSb,
        ln_g, ln_b, W_o, b_o, idxL, idxH, Mbuf, Zl, Zh, ns, nc);

    // 5: exclusive cumprod chains along seq_len
    md_cumprod_kernel<<<512, 256, 0, stream>>>(Zl, Zh, pNum, pLen, nc);

    // 6: gather + final product
    md_final_kernel<<<(int)(((long)ns * nc + 255) / 256), 256, 0, stream>>>(
        Mbuf, Zl, Zh, idxL, idxH, (float*)d_out, ns, nc);
}
